// GraphConv_21045339751032
// MI455X (gfx1250) — matrix operand order repaired and checked by reading, not compiled, not run on hardware
//
#include <hip/hip_runtime.h>
#include <math.h>

// ---------------------------------------------------------------------------
// GCNConv (sym-norm, self-loops) + bias + BatchNorm1d(eval) + ReLU  for MI455X
//
// Pipeline (all on `stream`, graph-capture safe):
//   1. init_ws      : deg = 1 (self-loop), agg = 0
//   2. count_deg    : deg[col[e]] += 1           (global_atomic_add_f32)
//   3. dinv         : deg <- rsqrt(deg)
//   4. gemm_xw      : xw = x @ W^T  via v_wmma_f32_16x16x32_f16.
//                     W staged fp32 -> LDS with global_load_async_to_lds_b128,
//                     then converted once to padded f16 LDS layout.
//   5. scatter      : agg[col] += dinv[row]*dinv[col] * xw[row]  (wave per edge)
//   6. epilogue     : relu(BN(agg + dinv^2*xw + bias))
// ---------------------------------------------------------------------------

typedef __attribute__((ext_vector_type(16))) _Float16 v16h;
typedef __attribute__((ext_vector_type(8)))  float    v8f;
typedef __attribute__((ext_vector_type(4)))  int      v4i;

typedef __attribute__((address_space(1))) v4i as1_v4i;   // global
typedef __attribute__((address_space(3))) v4i as3_v4i;   // LDS

#define D_IN  256
#define D_OUT 64
#define LDS_STRIDE 272   // 256 + 16 halfs pad: rows stay 32B-aligned, spreads banks

// ---------------------------------------------------------------- init ------
__global__ void init_ws_kernel(float* __restrict__ deg, float* __restrict__ agg,
                               int n, long long agg_total) {
    long long idx = (long long)blockIdx.x * blockDim.x + threadIdx.x;
    if (idx < n)         deg[idx] = 1.0f;   // self-loop contributes 1 to degree
    if (idx < agg_total) agg[idx] = 0.0f;
}

// ---------------------------------------------------------------- degree ----
__global__ void count_deg_kernel(const int* __restrict__ ecol,
                                 float* __restrict__ deg, int n_edges) {
    int e = blockIdx.x * blockDim.x + threadIdx.x;
    if (e < n_edges) atomicAdd(&deg[ecol[e]], 1.0f);
}

__global__ void dinv_kernel(float* __restrict__ deg, int n) {
    int i = blockIdx.x * blockDim.x + threadIdx.x;
    if (i < n) deg[i] = rsqrtf(deg[i]);     // deg >= 1 always (self-loops)
}

// ---------------------------------------------------------------- GEMM ------
// xw[n,64] = x[n,256] @ W[64,256]^T.  8 waves/block, 16 rows per wave,
// wave computes 16x64 via 4 column tiles x 8 K-chunks of 32.
__global__ void __launch_bounds__(256)
gemm_xw_kernel(const float* __restrict__ x, const float* __restrict__ W,
               float* __restrict__ xw, int n)
{
    __shared__ __align__(16) float    ldsWF[D_OUT * D_IN];        // 64 KB fp32 stage
    __shared__ __align__(32) _Float16 ldsW [D_OUT * LDS_STRIDE];  // 34 KB f16 padded

    const int tid = threadIdx.x;

    // --- Async DMA: W (64x256 fp32, 64 KB) global -> LDS, 16 x b128 per thread.
    {
        #pragma unroll
        for (int j = 0; j < 16; ++j) {
            int off = (tid + j * 256) * 4;           // float index, 16B granular
            __builtin_amdgcn_global_load_async_to_lds_b128(
                (as1_v4i*)(W + off), (as3_v4i*)&ldsWF[off], 0, 0);
        }
        __builtin_amdgcn_s_wait_asynccnt(0);
    }
    __syncthreads();

    // --- Convert once: LDS fp32 -> LDS f16 with padded row stride.
    {
        int row   = tid >> 2;            // 0..63
        int kbase = (tid & 3) * 64;      // 0,64,128,192
        const float* src = &ldsWF[row * D_IN + kbase];
        _Float16*    dst = &ldsW [row * LDS_STRIDE + kbase];
        #pragma unroll
        for (int j = 0; j < 64; j += 4) {
            float4 f = *(const float4*)(src + j);
            dst[j + 0] = (_Float16)f.x;
            dst[j + 1] = (_Float16)f.y;
            dst[j + 2] = (_Float16)f.z;
            dst[j + 3] = (_Float16)f.w;
        }
    }
    __syncthreads();

    const int wave    = tid >> 5;        // 0..7
    const int lane    = tid & 31;
    const int half    = lane >> 4;       // lane group: 0 or 1
    const int m       = lane & 15;
    const int rowBase = (blockIdx.x * 8 + wave) * 16;
    const int r       = rowBase + m;
    const int rc      = (r < n) ? r : (n - 1);   // clamp: branch-free A loads
    const float* xr   = x + (size_t)rc * D_IN;

    v8f c[4] = { v8f{}, v8f{}, v8f{}, v8f{} };

    #pragma unroll
    for (int k0 = 0; k0 < D_IN; k0 += 32) {
        // A fragment (16x32 f16): lanes 0-15 hold K=k0+0..15 of row m,
        // lanes 16-31 hold K=k0+16..31  (same K order as B).
        const float4* p = (const float4*)(xr + k0 + half * 16);
        float4 f0 = p[0], f1 = p[1], f2 = p[2], f3 = p[3];
        v16h a;
        a[0]=(_Float16)f0.x;  a[1]=(_Float16)f0.y;  a[2]=(_Float16)f0.z;  a[3]=(_Float16)f0.w;
        a[4]=(_Float16)f1.x;  a[5]=(_Float16)f1.y;  a[6]=(_Float16)f1.z;  a[7]=(_Float16)f1.w;
        a[8]=(_Float16)f2.x;  a[9]=(_Float16)f2.y;  a[10]=(_Float16)f2.z; a[11]=(_Float16)f2.w;
        a[12]=(_Float16)f3.x; a[13]=(_Float16)f3.y; a[14]=(_Float16)f3.z; a[15]=(_Float16)f3.w;

        // B fragments (32x16 f16): lane = output column within tile,
        // lanes 0-15 hold K=k0+0..15, lanes 16-31 hold K=k0+16..31.
        const int kb = k0 + half * 16;   // 32B-aligned offset into padded LDS row
        #pragma unroll
        for (int t = 0; t < 4; ++t) {
            v16h b = *(const v16h*)&ldsW[(t * 16 + m) * LDS_STRIDE + kb];
            c[t] = __builtin_amdgcn_wmma_f32_16x16x32_f16(
                       /*neg_a=*/false, a, /*neg_b=*/false, b,
                       /*c_mod=*/(short)0, c[t],
                       /*reuse_a=*/false, /*reuse_b=*/false);
        }
    }

    // C/D layout: VGPR j -> row (rowBase + half*8 + j), col = t*16 + m.
    #pragma unroll
    for (int t = 0; t < 4; ++t) {
        #pragma unroll
        for (int j = 0; j < 8; ++j) {
            int rr = rowBase + half * 8 + j;
            if (rr < n) xw[(size_t)rr * D_OUT + t * 16 + m] = c[t][j];
        }
    }
}

// ---------------------------------------------------------------- scatter ---
// One wave per edge: 32 lanes x float2 = 64 channels. Coalesced 256B gather of
// xw[row], 64 global_atomic_add_f32 into agg[col] (xw+agg are L2-resident).
__global__ void __launch_bounds__(256)
scatter_kernel(const int* __restrict__ erow, const int* __restrict__ ecol,
               const float* __restrict__ dinv, const float* __restrict__ xw,
               float* __restrict__ agg, int n_edges)
{
    int gw   = (int)(((long long)blockIdx.x * blockDim.x + threadIdx.x) >> 5);
    int lane = threadIdx.x & 31;
    if (gw >= n_edges) return;
    int rsrc = erow[gw];
    int cdst = ecol[gw];
    float norm = dinv[rsrc] * dinv[cdst];
    float2 v = ((const float2*)(xw + (size_t)rsrc * D_OUT))[lane];
    float* dst = agg + (size_t)cdst * D_OUT + lane * 2;
    atomicAdd(dst,     norm * v.x);
    atomicAdd(dst + 1, norm * v.y);
}

// ---------------------------------------------------------------- epilogue --
__global__ void epilogue_kernel(const float* __restrict__ agg, const float* __restrict__ xw,
                                const float* __restrict__ dinv, const float* __restrict__ bias,
                                const float* __restrict__ gamma, const float* __restrict__ beta,
                                const float* __restrict__ mean, const float* __restrict__ var,
                                float* __restrict__ out, long long total)
{
    long long idx = (long long)blockIdx.x * blockDim.x + threadIdx.x;
    if (idx >= total) return;
    int node = (int)(idx >> 6);
    int d    = (int)(idx & 63);
    float di = dinv[node];
    float v  = agg[idx] + di * di * xw[idx] + bias[d];            // + self-loop
    v = (v - mean[d]) * rsqrtf(var[d] + 1e-5f) * gamma[d] + beta[d];
    out[idx] = fmaxf(v, 0.0f);
}

// ---------------------------------------------------------------- launch ----
extern "C" void kernel_launch(void* const* d_in, const int* in_sizes, int n_in,
                              void* d_out, int out_size, void* d_ws, size_t ws_size,
                              hipStream_t stream)
{
    const float* x     = (const float*)d_in[0];
    const int*   eidx  = (const int*)  d_in[1];
    const float* W     = (const float*)d_in[2];
    const float* bias  = (const float*)d_in[3];
    const float* gamma = (const float*)d_in[4];
    const float* beta  = (const float*)d_in[5];
    const float* rmean = (const float*)d_in[6];
    const float* rvar  = (const float*)d_in[7];

    const int n       = in_sizes[0] / D_IN;
    const int n_edges = in_sizes[1] / 2;
    const int* erow = eidx;              // edge_index[0] = source
    const int* ecol = eidx + n_edges;    // edge_index[1] = target

    float* deg = (float*)d_ws;                     // n floats (becomes dinv)
    float* xw  = deg + n;                          // n*64 floats
    float* agg = xw + (size_t)n * D_OUT;           // n*64 floats
    float* out = (float*)d_out;

    const int T = 256;
    const long long agg_total = (long long)n * D_OUT;

    init_ws_kernel<<<(int)((agg_total + T - 1) / T), T, 0, stream>>>(deg, agg, n, agg_total);
    count_deg_kernel<<<(n_edges + T - 1) / T, T, 0, stream>>>(ecol, deg, n_edges);
    dinv_kernel<<<(n + T - 1) / T, T, 0, stream>>>(deg, n);
    gemm_xw_kernel<<<(n + 127) / 128, T, 0, stream>>>(x, W, xw, n);
    long long sthreads = (long long)n_edges * 32;
    scatter_kernel<<<(int)((sthreads + T - 1) / T), T, 0, stream>>>(erow, ecol, deg, xw, agg, n_edges);
    epilogue_kernel<<<(int)((agg_total + T - 1) / T), T, 0, stream>>>(agg, xw, deg, bias, gamma,
                                                                     beta, rmean, rvar, out, agg_total);
}
